// DecouplingFlowLayer_69295002353776
// MI455X (gfx1250) — matrix operand order repaired and checked by reading, not compiled, not run on hardware
//
#include <hip/hip_runtime.h>

typedef __attribute__((ext_vector_type(2)))  float  v2f;
typedef __attribute__((ext_vector_type(8)))  float  v8f;
typedef __attribute__((ext_vector_type(2)))  __bf16 v2bf;
typedef __attribute__((ext_vector_type(8)))  __bf16 v8bf;
typedef __attribute__((ext_vector_type(16))) __bf16 v16bf;

#define F_DIM  362            // K
#define KPAD   384            // K padded to 12 chunks of 32
#define KSTR   392            // LDS row stride (bf16 elems): 196 dwords = 4 mod 64 banks
#define N_DIM  512
#define S_DIM  128
#define MTOT   (4*128*512)    // 262144 GEMM rows
#define MTILE  32
#define NTILES (MTOT/MTILE)   // 8192
#define NBLK   512

// Split fp32 into bf16 hi/lo: a ~= hi + lo, residual ~2^-16 relative.
__device__ __forceinline__ __bf16 bhi(float f) { return (__bf16)f; }
__device__ __forceinline__ __bf16 blo(float f, __bf16 h) { return (__bf16)(f - (float)h); }

// A fragment (16-bit 16x32, ISA 7.12.2): lane<16 -> K{kc..kc+7} + K{kc+16..23};
// lane>=16 -> +8. Two 16B LDS loads, concatenated.
__device__ __forceinline__ v16bf ldfragA(const unsigned short* rowbase, int kc, int hi) {
    v8bf p0 = *(const v8bf*)(rowbase + kc + 8 * hi);
    v8bf p1 = *(const v8bf*)(rowbase + kc + 16 + 8 * hi);
    return __builtin_shufflevector(p0, p1, 0,1,2,3,4,5,6,7,8,9,10,11,12,13,14,15);
}
// B fragment (32x16): lane<16 -> K{kc..kc+7} + K{kc+16..23}; lane>=16 -> +8.
__device__ __forceinline__ v16bf ldfragB(const unsigned short* rowbase, int kc, int hi) {
    v8bf p0 = *(const v8bf*)(rowbase + kc + 8 * hi);
    v8bf p1 = *(const v8bf*)(rowbase + kc + 16 + 8 * hi);
    return __builtin_shufflevector(p0, p1, 0,1,2,3,4,5,6,7,8,9,10,11,12,13,14,15);
}

#define WMMA_BF16(A, B, C) \
    __builtin_amdgcn_wmma_f32_16x16x32_bf16(false, (A), false, (B), (short)0, (C), false, false)

__global__ __launch_bounds__(256) void dfl_wmma_bf16x3_kernel(
    const float* __restrict__ x,
    const float* __restrict__ Wg_w, const float* __restrict__ Wg_b,
    const float* __restrict__ Wh_w, const float* __restrict__ Wh_b,
    float* __restrict__ out)
{
    // bf16 hi/lo planes; weights resident for the whole (persistent) block.
    __shared__ unsigned short Whi[128 * KSTR];   // 100352 B
    __shared__ unsigned short Wlo[128 * KSTR];   // 100352 B
    __shared__ unsigned short Ahi[MTILE * KSTR]; //  25088 B
    __shared__ unsigned short Alo[MTILE * KSTR]; //  25088 B
    __shared__ float dl[MTILE];                  // rank-1 deltas: X_l - r
    __shared__ float dh[MTILE];                  //               X_h - r

    const int tid  = threadIdx.x;
    const int wave = tid >> 5;
    const int lane = tid & 31;
    const int l15  = lane & 15;
    const int hi   = lane >> 4;

    // ---- stage combined weights [Wg;Wh] -> bf16 hi/lo (zero-padded K>=362) --
    for (int rr = 0; rr < 16; ++rr) {
        const int jrow = wave * 16 + rr;
        const float* src = (jrow < 64) ? (Wg_w + (size_t)jrow * F_DIM)
                                       : (Wh_w + (size_t)(jrow - 64) * F_DIM);
        unsigned short* dsth = Whi + jrow * KSTR;
        unsigned short* dstl = Wlo + jrow * KSTR;
        for (int c2 = lane; c2 < KPAD / 2; c2 += 32) {
            v2f v = {0.f, 0.f};
            if (c2 < F_DIM / 2) v = *(const v2f*)(src + 2 * c2);
            const __bf16 hx = bhi(v.x), hy = bhi(v.y);
            v2bf vh = {hx, hy};
            v2bf vl = {blo(v.x, hx), blo(v.y, hy)};
            *(v2bf*)(dsth + 2 * c2) = vh;
            *(v2bf*)(dstl + 2 * c2) = vl;
        }
    }

    // Per-lane invariants (fp32-exact bias and feature-0 column from global).
    const int   j     = wave * 16 + l15;   // combined output column 0..127
    const float biasv = (j < 64) ? Wg_b[j] : Wh_b[j - 64];
    const float w0    = (j < 64) ? Wg_w[(size_t)j * F_DIM]
                                 : Wh_w[(size_t)(j - 64) * F_DIM];
    float* const outb = (j < 64) ? (out + j)
                                 : (out + (size_t)MTOT * 64 + (j - 64));
    __syncthreads();

    // ---- register double buffer for the A tile (wave w owns rows w+8t) ----
    v2f   abuf[4][6];
    float rc = 0.f, rp = 0.f;

    int tile = blockIdx.x;
    {   // prologue prefetch
        const int mbase = tile * MTILE;
        const int s     = (mbase / N_DIM) % S_DIM;
        const int pofs  = (s & 1) ? -N_DIM : N_DIM;
#pragma unroll
        for (int r4 = 0; r4 < 4; ++r4) {
            const float* src = x + (size_t)(mbase + wave + 8 * r4) * F_DIM;
#pragma unroll
            for (int i = 0; i < 6; ++i) {
                const int c2 = lane + 32 * i;
                v2f v = {0.f, 0.f};
                if (c2 < F_DIM / 2) v = *(const v2f*)(src + 2 * c2);
                abuf[r4][i] = v;
            }
        }
        if (tid < MTILE) {
            rc = x[(size_t)(mbase + tid) * F_DIM];
            rp = x[(size_t)(mbase + tid + pofs) * F_DIM];
        }
    }

    for (; tile < NTILES; tile += NBLK) {
        const int mbase = tile * MTILE;

        // ---- phase 1: regs -> LDS (bf16 split) + rank-1 deltas -------------
#pragma unroll
        for (int r4 = 0; r4 < 4; ++r4) {
            const int row = wave + 8 * r4;
            unsigned short* ah = Ahi + row * KSTR;
            unsigned short* al = Alo + row * KSTR;
#pragma unroll
            for (int i = 0; i < 6; ++i) {
                const int c2 = lane + 32 * i;
                const float fx = abuf[r4][i].x, fy = abuf[r4][i].y;
                const __bf16 hx = bhi(fx), hy = bhi(fy);
                v2bf vh = {hx, hy};
                v2bf vl = {blo(fx, hx), blo(fy, hy)};
                *(v2bf*)(ah + 2 * c2) = vh;
                *(v2bf*)(al + 2 * c2) = vl;
            }
        }
        if (tid < MTILE) {            // X_l-r = (rp-rc)/2 ; X_h-r = -(rc+rp)/2
            dl[tid] = 0.5f * (rp - rc);
            dh[tid] = -0.5f * (rp + rc);
        }
        __syncthreads();

        // ---- phase 2: prefetch next tile into regs (hidden under WMMA) ----
        const int nt = tile + NBLK;
        if (nt < NTILES) {
            const int nmb  = nt * MTILE;
            const int ns   = (nmb / N_DIM) % S_DIM;
            const int npof = (ns & 1) ? -N_DIM : N_DIM;
#pragma unroll
            for (int r4 = 0; r4 < 4; ++r4) {
                const float* src = x + (size_t)(nmb + wave + 8 * r4) * F_DIM;
#pragma unroll
                for (int i = 0; i < 6; ++i) {
                    const int c2 = lane + 32 * i;
                    v2f v = {0.f, 0.f};
                    if (c2 < F_DIM / 2) v = *(const v2f*)(src + 2 * c2);
                    abuf[r4][i] = v;
                }
            }
            if (tid < MTILE) {
                rc = x[(size_t)(nmb + tid) * F_DIM];
                rp = x[(size_t)(nmb + tid + npof) * F_DIM];
            }
        }

        // ---- phase 3: bf16x3 WMMA over K; wave = 32(M) x 16(N) ------------
        const unsigned short* a0h = Ahi + l15 * KSTR;          // rows 0..15
        const unsigned short* a0l = Alo + l15 * KSTR;
        const unsigned short* a1h = Ahi + (l15 + 16) * KSTR;   // rows 16..31
        const unsigned short* a1l = Alo + (l15 + 16) * KSTR;
        const unsigned short* bh_ = Whi + j * KSTR;
        const unsigned short* bl_ = Wlo + j * KSTR;

        v8f acc0 = {}, acc1 = {};
#pragma unroll 2
        for (int kc = 0; kc < KPAD; kc += 32) {
            v16bf bh  = ldfragB(bh_, kc, hi);
            v16bf bl  = ldfragB(bl_, kc, hi);
            v16bf ah0 = ldfragA(a0h, kc, hi);
            v16bf al0 = ldfragA(a0l, kc, hi);
            v16bf ah1 = ldfragA(a1h, kc, hi);
            v16bf al1 = ldfragA(a1l, kc, hi);
            acc0 = WMMA_BF16(ah0, bh, acc0);
            acc0 = WMMA_BF16(ah0, bl, acc0);
            acc0 = WMMA_BF16(al0, bh, acc0);
            acc1 = WMMA_BF16(ah1, bh, acc1);
            acc1 = WMMA_BF16(ah1, bl, acc1);
            acc1 = WMMA_BF16(al1, bh, acc1);
        }

        // ---- epilogue: bias + delta*W[:,0], fp32 store --------------------
        const float* dsel = (j < 64) ? dl : dh;
#pragma unroll
        for (int v = 0; v < 8; ++v) {
            const int row0 = v + 8 * hi;
            outb[(size_t)(mbase + row0) * 64] = acc0[v] + biasv + dsel[row0] * w0;
            const int row1 = row0 + 16;
            outb[(size_t)(mbase + row1) * 64] = acc1[v] + biasv + dsel[row1] * w0;
        }
        __syncthreads();   // protect Ahi/Alo/dl/dh before next staging
    }
}

extern "C" void kernel_launch(void* const* d_in, const int* in_sizes, int n_in,
                              void* d_out, int out_size, void* d_ws, size_t ws_size,
                              hipStream_t stream) {
    const float* x    = (const float*)d_in[0];
    const float* Wg_w = (const float*)d_in[1];
    const float* Wg_b = (const float*)d_in[2];
    const float* Wh_w = (const float*)d_in[3];
    const float* Wh_b = (const float*)d_in[4];
    float* out = (float*)d_out;

    dfl_wmma_bf16x3_kernel<<<dim3(NBLK), dim3(256), 0, stream>>>(
        x, Wg_w, Wg_b, Wh_w, Wh_b, out);
}
